// DeeperGCN_75136157876973
// MI455X (gfx1250) — matrix operand order repaired and checked by reading, not compiled, not run on hardware
//
#include <hip/hip_runtime.h>
#include <hip/hip_bf16.h>

#define EPS_GEN 1e-7f
#define BN_EPS  1e-5f

typedef __attribute__((ext_vector_type(16))) _Float16 v16h;
typedef __attribute__((ext_vector_type(8)))  _Float16 v8h;
typedef __attribute__((ext_vector_type(8)))  float    v8f;

// ---------------------------------------------------------------------------
// Atom encoder: h[n,d] = sum_{i<9} atom_emb[i][x[n,i]][d]
// ---------------------------------------------------------------------------
__global__ void k_atom(const int* __restrict__ x, const float* __restrict__ emb,
                       float* __restrict__ h, int N, int D) {
  int i = blockIdx.x * blockDim.x + threadIdx.x;
  if (i >= N * D) return;
  int n = i / D, d = i - n * D;
  const int* xr = x + 9 * n;
  float s = 0.f;
#pragma unroll
  for (int t = 0; t < 9; ++t)
    s += emb[((size_t)t * 119 + xr[t]) * D + d];
  h[i] = s;
}

__global__ void k_fill(float* __restrict__ p, float v, int n) {
  int i = blockIdx.x * blockDim.x + threadIdx.x;
  if (i < n) p[i] = v;
}

// ---------------------------------------------------------------------------
// Edge message helper: msg = relu(h[src] + bond_emb_sum) + eps ; scaled = msg*t
// ---------------------------------------------------------------------------
__device__ __forceinline__ float4 edge_scaled(
    const float* __restrict__ hin, const int* __restrict__ src,
    const int* __restrict__ ea, const float* __restrict__ bond,
    int e, int d, int D, float t, float4* msg_out) {
  int s  = src[e];
  int a0 = ea[3 * e + 0], a1 = ea[3 * e + 1], a2 = ea[3 * e + 2];
  float4 hv = *(const float4*)(hin  + (size_t)s * D + d);
  float4 b0 = *(const float4*)(bond + (size_t)(a0)      * D + d);
  float4 b1 = *(const float4*)(bond + (size_t)(6  + a1) * D + d);
  float4 b2 = *(const float4*)(bond + (size_t)(12 + a2) * D + d);
  float4 msg, sc;
  msg.x = fmaxf(hv.x + b0.x + b1.x + b2.x, 0.f) + EPS_GEN; sc.x = msg.x * t;
  msg.y = fmaxf(hv.y + b0.y + b1.y + b2.y, 0.f) + EPS_GEN; sc.y = msg.y * t;
  msg.z = fmaxf(hv.z + b0.z + b1.z + b2.z, 0.f) + EPS_GEN; sc.z = msg.z * t;
  msg.w = fmaxf(hv.w + b0.w + b1.w + b2.w, 0.f) + EPS_GEN; sc.w = msg.w * t;
  *msg_out = msg;
  return sc;
}

__device__ __forceinline__ float neg_inf_guard(float v) {
  return (v > -1e37f) ? v : 0.f;   // reference: where(isfinite(mx), mx, 0)
}

__global__ void k_edge_max(const float* __restrict__ hin, const int* __restrict__ src,
                           const int* __restrict__ dst, const int* __restrict__ ea,
                           const float* __restrict__ bond, const float* __restrict__ tptr,
                           float* __restrict__ mx, int E, int D) {
  int q = D >> 2, total = E * q;
  float t = *tptr;
  for (int i = blockIdx.x * blockDim.x + threadIdx.x; i < total;
       i += gridDim.x * blockDim.x) {
    int e = i / q, d = (i - e * q) << 2;
    float4 msg;
    float4 sc = edge_scaled(hin, src, ea, bond, e, d, D, t, &msg);
    float* p = mx + (size_t)dst[e] * D + d;
    atomicMax(p + 0, sc.x); atomicMax(p + 1, sc.y);
    atomicMax(p + 2, sc.z); atomicMax(p + 3, sc.w);
  }
}

__global__ void k_edge_expsum(const float* __restrict__ hin, const int* __restrict__ src,
                              const int* __restrict__ dst, const int* __restrict__ ea,
                              const float* __restrict__ bond, const float* __restrict__ tptr,
                              const float* __restrict__ mx, float* __restrict__ denom,
                              int E, int D) {
  int q = D >> 2, total = E * q;
  float t = *tptr;
  for (int i = blockIdx.x * blockDim.x + threadIdx.x; i < total;
       i += gridDim.x * blockDim.x) {
    int e = i / q, d = (i - e * q) << 2;
    float4 msg;
    float4 sc = edge_scaled(hin, src, ea, bond, e, d, D, t, &msg);
    size_t off = (size_t)dst[e] * D + d;
    float4 mv = *(const float4*)(mx + off);
    float* p = denom + off;
    atomicAdd(p + 0, __expf(sc.x - neg_inf_guard(mv.x)));
    atomicAdd(p + 1, __expf(sc.y - neg_inf_guard(mv.y)));
    atomicAdd(p + 2, __expf(sc.z - neg_inf_guard(mv.z)));
    atomicAdd(p + 3, __expf(sc.w - neg_inf_guard(mv.w)));
  }
}

__global__ void k_edge_agg(const float* __restrict__ hin, const int* __restrict__ src,
                           const int* __restrict__ dst, const int* __restrict__ ea,
                           const float* __restrict__ bond, const float* __restrict__ tptr,
                           const float* __restrict__ mx, const float* __restrict__ denom,
                           float* __restrict__ m, int E, int D) {
  int q = D >> 2, total = E * q;
  float t = *tptr;
  for (int i = blockIdx.x * blockDim.x + threadIdx.x; i < total;
       i += gridDim.x * blockDim.x) {
    int e = i / q, d = (i - e * q) << 2;
    float4 msg;
    float4 sc = edge_scaled(hin, src, ea, bond, e, d, D, t, &msg);
    size_t off = (size_t)dst[e] * D + d;
    float4 mv = *(const float4*)(mx + off);
    float4 dn = *(const float4*)(denom + off);
    float* p = m + off;
    atomicAdd(p + 0, msg.x * __expf(sc.x - neg_inf_guard(mv.x)) / fmaxf(dn.x, EPS_GEN));
    atomicAdd(p + 1, msg.y * __expf(sc.y - neg_inf_guard(mv.y)) / fmaxf(dn.y, EPS_GEN));
    atomicAdd(p + 2, msg.z * __expf(sc.z - neg_inf_guard(mv.z)) / fmaxf(dn.z, EPS_GEN));
    atomicAdd(p + 3, msg.w * __expf(sc.w - neg_inf_guard(mv.w)) / fmaxf(dn.w, EPS_GEN));
  }
}

__global__ void k_add(const float* __restrict__ a, const float* __restrict__ b,
                      float* __restrict__ out, int n) {
  int i = blockIdx.x * blockDim.x + threadIdx.x;
  if (i < n) out[i] = a[i] + b[i];
}

__global__ void k_resid(float* __restrict__ h, const float* __restrict__ s,
                        int n, int mode) {
  int i = blockIdx.x * blockDim.x + threadIdx.x;
  if (i < n) h[i] = mode ? (h[i] + s[i]) : s[i];
}

// ---------------------------------------------------------------------------
// BatchNorm batch statistics: one block per channel, LDS tree reduction.
// ---------------------------------------------------------------------------
__global__ void k_bn_stats(const float* __restrict__ x, int N, int C,
                           float* __restrict__ mu, float* __restrict__ var) {
  int c = blockIdx.x;
  float s = 0.f, sq = 0.f;
  for (int n = threadIdx.x; n < N; n += blockDim.x) {
    float v = x[(size_t)n * C + c];
    s += v; sq += v * v;
  }
  __shared__ float ss[256], ssq[256];
  ss[threadIdx.x] = s; ssq[threadIdx.x] = sq;
  __syncthreads();
  for (int off = blockDim.x >> 1; off > 0; off >>= 1) {
    if ((int)threadIdx.x < off) {
      ss[threadIdx.x]  += ss[threadIdx.x + off];
      ssq[threadIdx.x] += ssq[threadIdx.x + off];
    }
    __syncthreads();
  }
  if (threadIdx.x == 0) {
    float mval = ss[0] / (float)N;
    mu[c]  = mval;
    var[c] = ssq[0] / (float)N - mval * mval;
  }
}

__global__ void k_bn_apply(const float* __restrict__ y, const float* __restrict__ mu,
                           const float* __restrict__ var, const float* __restrict__ g,
                           const float* __restrict__ be, float* __restrict__ out,
                           int total, int C, int do_relu) {
  int i = blockIdx.x * blockDim.x + threadIdx.x;
  if (i >= total) return;
  int c = i & (C - 1);  // C is a power of two (128 / 256)
  float v = (y[i] - mu[c]) * rsqrtf(var[c] + BN_EPS) * g[c] + be[c];
  out[i] = do_relu ? fmaxf(v, 0.f) : v;
}

// ---------------------------------------------------------------------------
// Weight pre-pack: f32 [Kin,Kout] -> f16 in exact WMMA B-fragment order.
// Packed element i:  tile = i>>9 (= (k0/32)*(Kout/16) + ct), lane = (i>>4)&31,
// j = i&15 ;  k = k0 + 16*(lane>>4) + j ; col = 16*ct + (lane&15).
// blockIdx.y = layer.
// ---------------------------------------------------------------------------
__global__ void k_pack(const float* __restrict__ W, _Float16* __restrict__ Wp,
                       int Kin, int Kout) {
  int total = Kin * Kout;
  int l = blockIdx.y;
  const float* Ws = W + (size_t)l * total;
  _Float16*   Wd  = Wp + (size_t)l * total;
  int i = blockIdx.x * blockDim.x + threadIdx.x;
  if (i >= total) return;
  int j    = i & 15;
  int lane = (i >> 4) & 31;
  int tile = i >> 9;
  int nct  = Kout >> 4;
  int k0   = (tile / nct) << 5;
  int ct   = tile - (tile / nct) * nct;
  int k    = k0 + ((lane >> 4) << 4) + j;
  int col  = (ct << 4) + (lane & 15);
  Wd[i] = (_Float16)Ws[(size_t)k * Kout + col];
}

// ---------------------------------------------------------------------------
// Templated GEMM + bias via v_wmma_f32_16x16x32_f16.
// Block = 4 waves. Block owns a 16-row strip of A (staged f16 in LDS once).
// Wave w owns KOUT/64 contiguous 16-col tiles (multi-accumulator).
// A frag: two ds 16B reads per K-step; B frag: one contiguous 32B read of
// pre-packed f16 weights. K loop fully unrolled -> dense v_wmma stream.
// ---------------------------------------------------------------------------
template <int KIN, int KOUT>
__global__ void k_gemm_t(const float* __restrict__ A, const _Float16* __restrict__ Wp,
                         const float* __restrict__ bias, float* __restrict__ Y, int N) {
  constexpr int NT = KOUT / 64;          // col tiles per wave
  __shared__ _Float16 As[16 * KIN];

  int tid  = threadIdx.x;
  int row0 = blockIdx.x * 16;
  int lane = tid & 31, wave = tid >> 5;
  int r = lane & 15, half = lane >> 4;

  // prefetch this wave's weight panel (gfx1250 global_prefetch path)
  __builtin_prefetch(Wp + ((size_t)(wave * NT) << 9), 0, 0);

  // stage A strip (f32 -> f16) into LDS, coalesced
  for (int idx = tid; idx < 16 * KIN; idx += 128) {
    int rr = idx / KIN, kk = idx - rr * KIN;
    int grow = row0 + rr;
    As[idx] = (_Float16)(grow < N ? A[(size_t)grow * KIN + kk] : 0.f);
  }
  __syncthreads();

  v8f zero = {};
  v8f acc[NT];
#pragma unroll
  for (int t = 0; t < NT; ++t) acc[t] = zero;

#pragma unroll
  for (int k0 = 0; k0 < KIN; k0 += 32) {
    v8h alo = *(const v8h*)(As + r * KIN + k0 + half * 16);
    v8h ahi = *(const v8h*)(As + r * KIN + k0 + half * 16 + 8);
    v16h a;
#pragma unroll
    for (int j = 0; j < 8; ++j) { a[j] = alo[j]; a[j + 8] = ahi[j]; }
#pragma unroll
    for (int t = 0; t < NT; ++t) {
      int tile = (k0 >> 5) * (KOUT >> 4) + wave * NT + t;
      v16h b = *(const v16h*)(Wp + ((size_t)tile << 9) + lane * 16);
      acc[t] = __builtin_amdgcn_wmma_f32_16x16x32_f16(
          false, a, false, b, (short)0, acc[t], false, false);
    }
  }

#pragma unroll
  for (int t = 0; t < NT; ++t) {
    int col = (wave * NT + t) * 16 + r;
    float bv = bias[col];
    if (row0 + 16 <= N) {               // fast path (N % 16 == 0 in practice)
#pragma unroll
      for (int i = 0; i < 8; ++i)
        Y[(size_t)(row0 + half * 8 + i) * KOUT + col] = acc[t][i] + bv;
    } else {
#pragma unroll
      for (int i = 0; i < 8; ++i) {
        int rr = row0 + half * 8 + i;
        if (rr < N) Y[(size_t)rr * KOUT + col] = acc[t][i] + bv;
      }
    }
  }
}

// Generic fallback (runtime dims), same algorithm as round 1.
__global__ void k_gemm_generic(const float* __restrict__ A, const float* __restrict__ W,
                               const float* __restrict__ bias, float* __restrict__ Y,
                               int N, int Kin, int Kout) {
  int lane = threadIdx.x & 31;
  int wave = threadIdx.x >> 5;
  int row0 = blockIdx.x * 16;
  int col0 = (blockIdx.y * 4 + wave) * 16;
  if (col0 >= Kout) return;
  int r = lane & 15, half = lane >> 4;
  int row = row0 + r;
  bool rowok = row < N;
  const float* arow = A + (size_t)(rowok ? row : 0) * Kin;
  v8f acc = {};
  for (int k0 = 0; k0 < Kin; k0 += 32) {
    v16h a;
    int ka = k0 + half * 16;
#pragma unroll
    for (int j = 0; j < 8; ++j) {
      float a0 = arow[ka + j];
      float a1 = arow[ka + 8 + j];
      a[j]     = (_Float16)(rowok ? a0 : 0.f);
      a[j + 8] = (_Float16)(rowok ? a1 : 0.f);
    }
    v16h b;
    const float* wcol = W + (size_t)(k0 + half * 16) * Kout + (col0 + r);
#pragma unroll
    for (int j = 0; j < 16; ++j) b[j] = (_Float16)wcol[(size_t)j * Kout];
    acc = __builtin_amdgcn_wmma_f32_16x16x32_f16(false, a, false, b, (short)0, acc,
                                                 false, false);
  }
  int col = col0 + r;
  float bv = bias[col];
#pragma unroll
  for (int i = 0; i < 8; ++i) {
    int rr = row0 + half * 8 + i;
    if (rr < N) Y[(size_t)rr * Kout + col] = acc[i] + bv;
  }
}

// ---------------------------------------------------------------------------
// Global mean pool + prediction head
// ---------------------------------------------------------------------------
__global__ void k_pool_cnt(const int* __restrict__ batch, float* __restrict__ cnt, int N) {
  int n = blockIdx.x * blockDim.x + threadIdx.x;
  if (n < N) atomicAdd(&cnt[batch[n]], 1.f);
}

__global__ void k_pool_sum(const float* __restrict__ h, const int* __restrict__ batch,
                           float* __restrict__ hg, int N, int D) {
  int i = blockIdx.x * blockDim.x + threadIdx.x;
  if (i >= N * D) return;
  int n = i / D, d = i - n * D;
  atomicAdd(&hg[(size_t)batch[n] * D + d], h[i]);
}

__global__ void k_pred(const float* __restrict__ hg, const float* __restrict__ cnt,
                       const float* __restrict__ pW, const float* __restrict__ pb,
                       float* __restrict__ out, int G, int D) {
  int g = blockIdx.x * blockDim.x + threadIdx.x;
  if (g >= G) return;
  float c = fmaxf(cnt[g], 1.f);
  float acc = pb[0];
  const float* row = hg + (size_t)g * D;
  for (int d = 0; d < D; ++d) acc += (row[d] / c) * pW[d];
  out[g] = acc;
}

// ---------------------------------------------------------------------------
// Host orchestration
// ---------------------------------------------------------------------------
extern "C" void kernel_launch(void* const* d_in, const int* in_sizes, int n_in,
                              void* d_out, int out_size, void* d_ws, size_t ws_size,
                              hipStream_t stream) {
  const int*   x         = (const int*)  d_in[0];
  const int*   edge_idx  = (const int*)  d_in[1];
  const int*   edge_attr = (const int*)  d_in[2];
  const int*   batch     = (const int*)  d_in[3];
  const float* atom_emb  = (const float*)d_in[4];
  const float* bond_emb  = (const float*)d_in[5];
  const float* W1 = (const float*)d_in[6];  const float* b1 = (const float*)d_in[7];
  const float* g1 = (const float*)d_in[8];  const float* be1= (const float*)d_in[9];
  const float* W2 = (const float*)d_in[10]; const float* b2 = (const float*)d_in[11];
  const float* g2 = (const float*)d_in[12]; const float* be2= (const float*)d_in[13];
  const float* W3 = (const float*)d_in[14]; const float* b3 = (const float*)d_in[15];
  const float* t_arr  = (const float*)d_in[16];
  const float* norm_g = (const float*)d_in[17];
  const float* norm_b = (const float*)d_in[18];
  const float* predW  = (const float*)d_in[19];
  const float* predb  = (const float*)d_in[20];
  float* out = (float*)d_out;

  const int L = in_sizes[16];          // 20
  const int N = in_sizes[0] / 9;       // 100000
  const int E = in_sizes[1] / 2;       // 400000
  const int D = in_sizes[19];          // 128
  const int H = in_sizes[7] / L;       // 256
  const int G = out_size;              // 4096
  const bool fast = (D == 128 && H == 256);

  const int* src = edge_idx;
  const int* dst = edge_idx + E;

  // ---- workspace carve-out (256B aligned) ----
  char* wp = (char*)d_ws;
  auto alloc_b = [&](size_t bytes) {
    char* p = wp;
    wp += ((bytes + 255) / 256) * 256;
    return p;
  };
  float* h   = (float*)alloc_b((size_t)N * D * 4);
  float* h2  = (float*)alloc_b((size_t)N * D * 4);
  float* m   = (float*)alloc_b((size_t)N * D * 4);   // messages; reused as conv out
  float* hh  = (float*)alloc_b((size_t)N * D * 4);
  float* P   = (float*)alloc_b((size_t)N * H * 4);   // GEMM ping (doubles as mx)
  float* Q   = (float*)alloc_b((size_t)N * H * 4);   // GEMM pong (doubles as denom)
  float* mu  = (float*)alloc_b(256 * 4);
  float* var = (float*)alloc_b(256 * 4);
  float* hg  = (float*)alloc_b((size_t)G * D * 4);
  float* cnt = (float*)alloc_b((size_t)G * 4);
  _Float16* Wp1 = (_Float16*)alloc_b((size_t)L * D * H * 2);
  _Float16* Wp2 = (_Float16*)alloc_b((size_t)L * H * H * 2);
  _Float16* Wp3 = (_Float16*)alloc_b((size_t)L * H * D * 2);
  (void)ws_size; (void)n_in;

  const int ND = N * D, NH = N * H;
  const int TB = 256;
  auto cdiv = [](int a, int b) { return (a + b - 1) / b; };
  const int edge_blocks = cdiv(E * (D / 4), TB);

  // ---- pre-pack all layer weights into WMMA B-fragment order (f16) ----
  if (fast) {
    k_pack<<<dim3(cdiv(D * H, TB), L), TB, 0, stream>>>(W1, Wp1, D, H);
    k_pack<<<dim3(cdiv(H * H, TB), L), TB, 0, stream>>>(W2, Wp2, H, H);
    k_pack<<<dim3(cdiv(H * D, TB), L), TB, 0, stream>>>(W3, Wp3, H, D);
  }

  // ---- atom encoder ----
  k_atom<<<cdiv(ND, TB), TB, 0, stream>>>(x, atom_emb, h, N, D);

  for (int l = 0; l < L; ++l) {
    const float* hin;
    if (l == 0) {
      hin = h;
    } else {
      k_bn_stats<<<D, TB, 0, stream>>>(h, N, D, mu, var);
      k_bn_apply<<<cdiv(ND, TB), TB, 0, stream>>>(h, mu, var,
          norm_g + (size_t)(l - 1) * D, norm_b + (size_t)(l - 1) * D,
          h2, ND, D, /*relu=*/1);
      hin = h2;
    }

    // ---- scatter-softmax message aggregation ----
    const float* bond_l = bond_emb + (size_t)l * 3 * 6 * D;
    const float* t_l    = t_arr + l;
    float* mx    = P;
    float* denom = Q;
    k_fill<<<cdiv(ND, TB), TB, 0, stream>>>(mx, -__builtin_huge_valf(), ND);
    hipMemsetAsync(denom, 0, (size_t)ND * sizeof(float), stream);
    hipMemsetAsync(m,     0, (size_t)ND * sizeof(float), stream);

    k_edge_max   <<<edge_blocks, TB, 0, stream>>>(hin, src, dst, edge_attr, bond_l, t_l, mx, E, D);
    k_edge_expsum<<<edge_blocks, TB, 0, stream>>>(hin, src, dst, edge_attr, bond_l, t_l, mx, denom, E, D);
    k_edge_agg   <<<edge_blocks, TB, 0, stream>>>(hin, src, dst, edge_attr, bond_l, t_l, mx, denom, m, E, D);

    k_add<<<cdiv(ND, TB), TB, 0, stream>>>(hin, m, hh, ND);

    // ---- MLP: D->H (bn,relu) -> H (bn,relu) -> D ----
    if (fast) {
      k_gemm_t<128, 256><<<cdiv(N, 16), 128, 0, stream>>>(
          hh, Wp1 + (size_t)l * D * H, b1 + (size_t)l * H, P, N);
    } else {
      k_gemm_generic<<<dim3(cdiv(N, 16), cdiv(H, 64)), 128, 0, stream>>>(
          hh, W1 + (size_t)l * D * H, b1 + (size_t)l * H, P, N, D, H);
    }
    k_bn_stats<<<H, TB, 0, stream>>>(P, N, H, mu, var);
    k_bn_apply<<<cdiv(NH, TB), TB, 0, stream>>>(P, mu, var, g1 + (size_t)l * H,
                                                be1 + (size_t)l * H, Q, NH, H, 1);

    if (fast) {
      k_gemm_t<256, 256><<<cdiv(N, 16), 128, 0, stream>>>(
          Q, Wp2 + (size_t)l * H * H, b2 + (size_t)l * H, P, N);
    } else {
      k_gemm_generic<<<dim3(cdiv(N, 16), cdiv(H, 64)), 128, 0, stream>>>(
          Q, W2 + (size_t)l * H * H, b2 + (size_t)l * H, P, N, H, H);
    }
    k_bn_stats<<<H, TB, 0, stream>>>(P, N, H, mu, var);
    k_bn_apply<<<cdiv(NH, TB), TB, 0, stream>>>(P, mu, var, g2 + (size_t)l * H,
                                                be2 + (size_t)l * H, Q, NH, H, 1);

    if (fast) {
      k_gemm_t<256, 128><<<cdiv(N, 16), 128, 0, stream>>>(
          Q, Wp3 + (size_t)l * H * D, b3 + (size_t)l * D, m, N);
    } else {
      k_gemm_generic<<<dim3(cdiv(N, 16), cdiv(D, 64)), 128, 0, stream>>>(
          Q, W3 + (size_t)l * H * D, b3 + (size_t)l * D, m, N, H, D);
    }

    k_resid<<<cdiv(ND, TB), TB, 0, stream>>>(h, m, ND, l == 0 ? 0 : 1);
  }

  // ---- final norm (no relu), mean pool, prediction head ----
  k_bn_stats<<<D, TB, 0, stream>>>(h, N, D, mu, var);
  k_bn_apply<<<cdiv(ND, TB), TB, 0, stream>>>(h, mu, var,
      norm_g + (size_t)(L - 1) * D, norm_b + (size_t)(L - 1) * D, h2, ND, D, 0);

  hipMemsetAsync(hg,  0, (size_t)G * D * sizeof(float), stream);
  hipMemsetAsync(cnt, 0, (size_t)G * sizeof(float), stream);
  k_pool_cnt<<<cdiv(N, TB), TB, 0, stream>>>(batch, cnt, N);
  k_pool_sum<<<cdiv(ND, TB), TB, 0, stream>>>(h2, batch, hg, N, D);
  k_pred<<<cdiv(G, TB), TB, 0, stream>>>(hg, cnt, predW, predb, out, G, D);
}
